// GNN_78718160601379
// MI455X (gfx1250) — matrix operand order repaired and checked by reading, not compiled, not run on hardware
//
#include <hip/hip_runtime.h>
#include <math.h>

#define N_CUSTN 100000
#define N_PRODN 20000
#define NEDGE   1000000
#define NEL     500000
#define DC   15
#define DPIN 769
#define PHN  512
#define HD   64
#define EDN  8

typedef __attribute__((ext_vector_type(16))) _Float16 v16h;
typedef __attribute__((ext_vector_type(8)))  _Float16 v8h;
typedef __attribute__((ext_vector_type(8)))  float    v8f;

#define BM 64
#define BN 64
#define BK 32

// ---------------------------------------------------------------------------
// Helpers
// ---------------------------------------------------------------------------
__device__ __forceinline__ float act_apply(float v, int act) {
  if (act == 1) return v > 0.f ? v : 0.f;          // relu
  if (act == 2) return v > 0.f ? v : 0.01f * v;    // leaky_relu(0.01)
  return v;
}

__device__ __forceinline__ void atomicMaxFloat(float* addr, float val) {
  // IEEE-monotonic int punning trick; init value must be -inf.
  if (val >= 0.f)
    atomicMax((int*)addr, __float_as_int(val));
  else
    atomicMin((unsigned int*)addr, __float_as_uint(val));
}

// Fragment-order LDS staging (see ISA 7.12.2 layouts):
// A (16x32 f16): m = w*16+(L&15), k = e + 16*(L>>4);
//   an aligned 8-wide k-group of a row = one contiguous lane half -> 16B store.
// B (32x16 f16): n = nt*16+(L&15), k = e + 16*(L>>4);
//   one (nt,L) = one column's 16 k -> two 16B stores.

// ---------------------------------------------------------------------------
// Generic f32-in / f32-out GEMM on the WMMA f16 path.
// C[M,N] = act(A[M,K] @ B[K,N] + bias), A/B row-major.
// Block = 128 threads (4 waves); wave w owns rows [w*16, w*16+16) x 64 cols.
// KVEC=true: K % 32 == 0 and A rows 16B-aligned -> float4 A staging, no
// k-guards. Masking is always multiply-by-{0,1} applied AFTER batched loads so
// the compiler never predicates the loads (no select-of-load).
// ---------------------------------------------------------------------------
template <bool KVEC>
__global__ __launch_bounds__(128)
void gemm_wmma_f16_t(const float* __restrict__ A, const float* __restrict__ B,
                     const float* __restrict__ bias, float* __restrict__ C,
                     int M, int N, int K, int act)
{
  __shared__ _Float16 AsF[4][32][16] __attribute__((aligned(32)));
  __shared__ _Float16 BsF[4][32][16] __attribute__((aligned(32)));
  const int row0 = blockIdx.x * BM;
  const int col0 = blockIdx.y * BN;
  const int tid  = threadIdx.x;
  const int lane = tid & 31;
  const int wave = tid >> 5;

  v8f acc[4] = {};

  for (int k0 = 0; k0 < K; k0 += BK) {
    // ---- Stage A (64x32): 256 groups of 8; thread handles 2 groups. ----
#pragma unroll
    for (int gg = 0; gg < 2; ++gg) {
      const int g  = tid + gg * 128;
      const int m  = g >> 2;        // 0..63
      const int kg = g & 3;         // k-group: k = kg*8 .. kg*8+7
      const int gr = row0 + m;
      const int gk = k0 + kg * 8;
      const int rc = gr < M ? gr : (M - 1);      // clamped row
      const float rmask = gr < M ? 1.f : 0.f;
      v8h hv;
      if (KVEC) {
        const float* ap = A + (size_t)rc * K + gk;   // 32B aligned
        const float4 a0 = ((const float4*)ap)[0];
        const float4 a1 = ((const float4*)ap)[1];
        hv[0] = (_Float16)(a0.x * rmask); hv[1] = (_Float16)(a0.y * rmask);
        hv[2] = (_Float16)(a0.z * rmask); hv[3] = (_Float16)(a0.w * rmask);
        hv[4] = (_Float16)(a1.x * rmask); hv[5] = (_Float16)(a1.y * rmask);
        hv[6] = (_Float16)(a1.z * rmask); hv[7] = (_Float16)(a1.w * rmask);
      } else {
        const float* arow = A + (size_t)rc * K;
        float vals[8];
#pragma unroll
        for (int j = 0; j < 8; ++j) {                // batched loads only
          const int kk = gk + j;
          const int kc = kk < K ? kk : (K - 1);
          vals[j] = arow[kc];
        }
#pragma unroll
        for (int j = 0; j < 8; ++j) {                // mask + convert after
          const float em = (gk + j < K) ? rmask : 0.f;
          hv[j] = (_Float16)(vals[j] * em);
        }
      }
      const int wv = m >> 4;
      const int L  = (m & 15) + ((kg >> 1) << 4);
      const int eb = (kg & 1) << 3;
      *(v8h*)&AsF[wv][L][eb] = hv;
    }
    // ---- Stage B (32x64): thread owns one (nt, L) fragment slot. ----
    {
      const int nt = tid >> 5;
      const int L  = tid & 31;
      const int n  = col0 + nt * 16 + (L & 15);
      const int nc = n < N ? n : (N - 1);
      const float nmask = n < N ? 1.f : 0.f;
      const int kb = k0 + ((L >> 4) << 4);
      float vals[16];
#pragma unroll
      for (int j = 0; j < 16; ++j) {                 // batched loads only
        const int gk = kb + j;
        const int kc = KVEC ? gk : (gk < K ? gk : (K - 1));
        vals[j] = B[(size_t)kc * N + nc];
      }
      v8h h0, h1;
#pragma unroll
      for (int j = 0; j < 16; ++j) {                 // mask + convert after
        const float em = KVEC ? nmask : ((kb + j < K) ? nmask : 0.f);
        const _Float16 hval = (_Float16)(vals[j] * em);
        if (j < 8) h0[j] = hval; else h1[j - 8] = hval;
      }
      *(v8h*)&BsF[nt][L][0] = h0;
      *(v8h*)&BsF[nt][L][8] = h1;
    }
    __syncthreads();

    const v16h af = *(const v16h*)&AsF[wave][lane][0];
#pragma unroll
    for (int nt = 0; nt < 4; ++nt) {
      const v16h bf = *(const v16h*)&BsF[nt][lane][0];
      acc[nt] = __builtin_amdgcn_wmma_f32_16x16x32_f16(
          false, af, false, bf, (short)0, acc[nt], false, false);
    }
    __syncthreads();
  }

  // Epilogue. C layout: lane holds col = lane%16 (+16*nt), rows r + 8*(lane/16).
#pragma unroll
  for (int nt = 0; nt < 4; ++nt) {
    int gc = col0 + nt * 16 + (lane & 15);
    if (gc >= N) continue;
    float bv = bias ? bias[gc] : 0.f;
#pragma unroll
    for (int r = 0; r < 8; ++r) {
      int gr = row0 + wave * 16 + ((lane >> 4) << 3) + r;
      if (gr < M)
        C[(size_t)gr * N + gc] = act_apply(acc[nt][r] + bv, act);
    }
  }
}

// ---------------------------------------------------------------------------
// Decoder, fully fused:
//   z = relu(concat(h_c[el_src], h_p[el_dst]) @ W1 + b1)   (K=128, N=64)
//   out = sigmoid(z @ W2 + b2)                              (fused epilogue)
// ---------------------------------------------------------------------------
__global__ __launch_bounds__(128)
void decoder_wmma(const float* __restrict__ hc, const float* __restrict__ hp,
                  const int* __restrict__ elsrc, const int* __restrict__ eldst,
                  const float* __restrict__ W1, const float* __restrict__ b1,
                  const float* __restrict__ W2, const float* __restrict__ b2,
                  float* __restrict__ out, int M)
{
  __shared__ _Float16 AsF[4][32][16] __attribute__((aligned(32)));
  __shared__ _Float16 BsF[4][32][16] __attribute__((aligned(32)));
  const int row0 = blockIdx.x * BM;
  const int tid  = threadIdx.x;
  const int lane = tid & 31;
  const int wave = tid >> 5;

  v8f acc[4] = {};

  for (int k0 = 0; k0 < 2 * HD; k0 += BK) {
    // ---- Stage gathered concat A rows, branch-free (row clamp + mask). ----
    // 8-wide k groups stay inside one source row (gk%64 multiple of 8);
    // workspace rows are 256B-aligned -> float4 loads are safe.
#pragma unroll
    for (int gg = 0; gg < 2; ++gg) {
      const int g  = tid + gg * 128;
      const int m  = g >> 2;
      const int kg = g & 3;
      const int gr = row0 + m;
      const int gk = k0 + kg * 8;
      const int rc = gr < M ? gr : (M - 1);
      const float msk = gr < M ? 1.f : 0.f;
      const float* srcp = (gk < HD)
          ? (hc + (size_t)elsrc[rc] * HD + gk)
          : (hp + (size_t)eldst[rc] * HD + (gk - HD));
      float4 a0 = ((const float4*)srcp)[0];
      float4 a1 = ((const float4*)srcp)[1];
      v8h hv;
      hv[0] = (_Float16)(a0.x * msk); hv[1] = (_Float16)(a0.y * msk);
      hv[2] = (_Float16)(a0.z * msk); hv[3] = (_Float16)(a0.w * msk);
      hv[4] = (_Float16)(a1.x * msk); hv[5] = (_Float16)(a1.y * msk);
      hv[6] = (_Float16)(a1.z * msk); hv[7] = (_Float16)(a1.w * msk);
      const int wv = m >> 4;
      const int L  = (m & 15) + ((kg >> 1) << 4);
      const int eb = (kg & 1) << 3;
      *(v8h*)&AsF[wv][L][eb] = hv;
    }
    // ---- Stage W1 (K=128, N=64 exact; no guards). ----
    {
      const int nt = tid >> 5;
      const int L  = tid & 31;
      const int n  = nt * 16 + (L & 15);
      const int kb = k0 + ((L >> 4) << 4);
      float vals[16];
#pragma unroll
      for (int j = 0; j < 16; ++j)
        vals[j] = W1[(size_t)(kb + j) * HD + n];
      v8h h0, h1;
#pragma unroll
      for (int j = 0; j < 16; ++j) {
        const _Float16 hval = (_Float16)vals[j];
        if (j < 8) h0[j] = hval; else h1[j - 8] = hval;
      }
      *(v8h*)&BsF[nt][L][0] = h0;
      *(v8h*)&BsF[nt][L][8] = h1;
    }
    __syncthreads();

    const v16h af = *(const v16h*)&AsF[wave][lane][0];
#pragma unroll
    for (int nt = 0; nt < 4; ++nt) {
      const v16h bf = *(const v16h*)&BsF[nt][lane][0];
      acc[nt] = __builtin_amdgcn_wmma_f32_16x16x32_f16(
          false, af, false, bf, (short)0, acc[nt], false, false);
    }
    __syncthreads();
  }

  // Fused second GEMM: per-lane partial over its 4 columns, then shuffle
  // reduce across the 16-lane group that shares the same 8 rows.
  float part[8] = {0.f, 0.f, 0.f, 0.f, 0.f, 0.f, 0.f, 0.f};
#pragma unroll
  for (int nt = 0; nt < 4; ++nt) {
    const int col = nt * 16 + (lane & 15);
    const float bv = b1[col];
    const float wv = W2[col];
#pragma unroll
    for (int r = 0; r < 8; ++r) {
      float z = acc[nt][r] + bv;
      z = z > 0.f ? z : 0.f;
      part[r] += z * wv;
    }
  }
#pragma unroll
  for (int off = 1; off < 16; off <<= 1) {
#pragma unroll
    for (int r = 0; r < 8; ++r)
      part[r] += __shfl_xor(part[r], off, 32);
  }
  if ((lane & 15) == 0) {
    const float b2v = b2[0];
#pragma unroll
    for (int r = 0; r < 8; ++r) {
      int gr = row0 + wave * 16 + ((lane >> 4) << 3) + r;
      if (gr < M)
        out[gr] = 1.f / (1.f + expf(-(part[r] + b2v)));
    }
  }
}

// ---------------------------------------------------------------------------
// Edge kernels (wave-per-edge; lane handles h = lane and h = lane + 32)
// ---------------------------------------------------------------------------
__global__ __launch_bounds__(256)
void edge_logit_max(const float* __restrict__ ea, const float* __restrict__ We,
                    const float* __restrict__ att,
                    const float* __restrict__ xl, const float* __restrict__ xr,
                    const int* __restrict__ src, const int* __restrict__ dst,
                    float* __restrict__ logit, float* __restrict__ mbuf, int E)
{
  const int wid  = (blockIdx.x * 256 + threadIdx.x) >> 5;
  const int lane = threadIdx.x & 31;
  if (wid >= E) return;
  const int s = src[wid], d = dst[wid];
  const float* e = ea + (size_t)wid * EDN;

  // The edge_attr stream is perfectly sequential across waves; pull the line
  // ~2KB ahead into cache (global_prefetch_b8, speculative).
  __builtin_prefetch(e + 64 * EDN, 0, 0);

  float w0 = 0.f, w1 = 0.f;
#pragma unroll
  for (int dd = 0; dd < EDN; ++dd) {
    float ev = e[dd];
    w0 += ev * We[dd * HD + lane];
    w1 += ev * We[dd * HD + lane + 32];
  }
  float f0 = xl[(size_t)s * HD + lane]      + xr[(size_t)d * HD + lane]      + w0;
  float f1 = xl[(size_t)s * HD + lane + 32] + xr[(size_t)d * HD + lane + 32] + w1;
  f0 = f0 > 0.f ? f0 : 0.2f * f0;
  f1 = f1 > 0.f ? f1 : 0.2f * f1;
  float p = f0 * att[lane] + f1 * att[lane + 32];
#pragma unroll
  for (int off = 16; off > 0; off >>= 1) p += __shfl_xor(p, off, 32);
  if (lane == 0) {
    logit[wid] = p;
    atomicMaxFloat(&mbuf[d], p);
  }
}

__global__ __launch_bounds__(256)
void edge_exp_sum(const float* __restrict__ logit, const float* __restrict__ mbuf,
                  const int* __restrict__ dst, float* __restrict__ ex,
                  float* __restrict__ den, int E)
{
  int i = blockIdx.x * 256 + threadIdx.x;
  if (i >= E) return;
  int d = dst[i];
  float v = expf(logit[i] - mbuf[d]);
  ex[i] = v;
  atomicAdd(&den[d], v);
}

__global__ __launch_bounds__(256)
void edge_scatter(const float* __restrict__ ex, const float* __restrict__ den,
                  const float* __restrict__ xl,
                  const int* __restrict__ src, const int* __restrict__ dst,
                  float* __restrict__ outp, int E)
{
  const int wid  = (blockIdx.x * 256 + threadIdx.x) >> 5;
  const int lane = threadIdx.x & 31;
  if (wid >= E) return;
  const int s = src[wid], d = dst[wid];
  const float alpha = ex[wid] / (den[d] + 1e-16f);
  atomicAdd(&outp[(size_t)d * HD + lane],      alpha * xl[(size_t)s * HD + lane]);
  atomicAdd(&outp[(size_t)d * HD + lane + 32], alpha * xl[(size_t)s * HD + lane + 32]);
}

__global__ __launch_bounds__(256)
void fill_kernel(float* __restrict__ p, size_t n, float v)
{
  size_t i = (size_t)blockIdx.x * 256 + threadIdx.x;
  if (i < n) p[i] = v;
}

__global__ __launch_bounds__(256)
void bias_act_kernel(float* __restrict__ outp, const float* __restrict__ bias,
                     size_t n, int relu)
{
  size_t i = (size_t)blockIdx.x * 256 + threadIdx.x;
  if (i >= n) return;
  float v = outp[i] + bias[i & (HD - 1)];
  if (relu) v = v > 0.f ? v : 0.f;
  outp[i] = v;
}

// ---------------------------------------------------------------------------
// Host-side orchestration
// ---------------------------------------------------------------------------
static inline unsigned cdiv(size_t a, size_t b) { return (unsigned)((a + b - 1) / b); }

extern "C" void kernel_launch(void* const* d_in, const int* in_sizes, int n_in,
                              void* d_out, int out_size, void* d_ws, size_t ws_size,
                              hipStream_t stream)
{
  const float* products   = (const float*)d_in[0];
  const float* x_customer = (const float*)d_in[1];
  const float* edge_attr  = (const float*)d_in[2];
  const float* pe_W1 = (const float*)d_in[3];
  const float* pe_b1 = (const float*)d_in[4];
  const float* pe_W2 = (const float*)d_in[5];
  const float* pe_b2 = (const float*)d_in[6];
  const float* l1bp_Wl  = (const float*)d_in[7];
  const float* l1bp_bl  = (const float*)d_in[8];
  const float* l1bp_Wr  = (const float*)d_in[9];
  const float* l1bp_br  = (const float*)d_in[10];
  const float* l1bp_We  = (const float*)d_in[11];
  const float* l1bp_att = (const float*)d_in[12];
  const float* l1bp_bias= (const float*)d_in[13];
  const float* l1pc_Wl  = (const float*)d_in[14];
  const float* l1pc_bl  = (const float*)d_in[15];
  const float* l1pc_Wr  = (const float*)d_in[16];
  const float* l1pc_br  = (const float*)d_in[17];
  const float* l1pc_We  = (const float*)d_in[18];
  const float* l1pc_att = (const float*)d_in[19];
  const float* l1pc_bias= (const float*)d_in[20];
  const float* Wl23   = (const float*)d_in[21];
  const float* bl23   = (const float*)d_in[22];
  const float* Wr23   = (const float*)d_in[23];
  const float* br23   = (const float*)d_in[24];
  const float* We23   = (const float*)d_in[25];
  const float* att23  = (const float*)d_in[26];
  const float* bias23 = (const float*)d_in[27];
  const float* dec_W1 = (const float*)d_in[28];
  const float* dec_b1 = (const float*)d_in[29];
  const float* dec_W2 = (const float*)d_in[30];
  const float* dec_b2 = (const float*)d_in[31];
  const int* ei_src = (const int*)d_in[32];
  const int* ei_dst = (const int*)d_in[33];
  const int* el_src = (const int*)d_in[34];
  const int* el_dst = (const int*)d_in[35];

  // Workspace carve-up (floats). Sequential phases allow XPH to alias XL/XR.
  float* ws = (float*)d_ws;
  size_t o = 0;
  float* XL  = ws + o; o += (size_t)N_CUSTN * HD;  // 6.4M
  float* XR  = ws + o; o += (size_t)N_CUSTN * HD;  // 6.4M
  float* XPH = XL;  // encoder hidden [20000*512] = 10.24M <= XL+XR (12.8M)
  float* XP  = ws + o; o += (size_t)N_PRODN * HD;
  float* LOG = ws + o; o += (size_t)NEDGE;
  float* EX  = ws + o; o += (size_t)NEDGE;
  float* MX  = ws + o; o += (size_t)N_CUSTN;
  float* DEN = ws + o; o += (size_t)N_CUSTN;
  float* HPa = ws + o; o += (size_t)N_PRODN * HD;
  float* HPb = ws + o; o += (size_t)N_PRODN * HD;
  float* HCa = ws + o; o += (size_t)N_CUSTN * HD;
  float* HCb = ws + o; o += (size_t)N_CUSTN * HD;
  (void)ws_size; (void)in_sizes; (void)n_in; (void)out_size;

  auto gemm = [&](const float* A, const float* B, const float* bias, float* C,
                  int M, int N, int K, int act) {
    dim3 g(cdiv(M, BM), cdiv(N, BN));
    if ((K & 31) == 0)   // K multiple of 32 -> vectorized A staging
      gemm_wmma_f16_t<true><<<g, dim3(128), 0, stream>>>(A, B, bias, C, M, N, K, act);
    else
      gemm_wmma_f16_t<false><<<g, dim3(128), 0, stream>>>(A, B, bias, C, M, N, K, act);
  };
  auto fill = [&](float* p, size_t n, float v) {
    fill_kernel<<<cdiv(n, 256), 256, 0, stream>>>(p, n, v);
  };
  auto gat = [&](const float* xsrc, int dsrc, int nsrc,
                 const float* xdst, int ddst, int ndst,
                 const int* src, const int* dst,
                 const float* Wl, const float* bl, const float* Wr, const float* br,
                 const float* We, const float* att, const float* bias,
                 float* outp, int relu) {
    gemm(xsrc, Wl, bl, XL, nsrc, HD, dsrc, 0);
    gemm(xdst, Wr, br, XR, ndst, HD, ddst, 0);
    fill(MX, (size_t)ndst, -INFINITY);
    fill(DEN, (size_t)ndst, 0.f);
    fill(outp, (size_t)ndst * HD, 0.f);
    edge_logit_max<<<cdiv((size_t)NEDGE * 32, 256), 256, 0, stream>>>(
        edge_attr, We, att, XL, XR, src, dst, LOG, MX, NEDGE);
    edge_exp_sum<<<cdiv(NEDGE, 256), 256, 0, stream>>>(LOG, MX, dst, EX, DEN, NEDGE);
    edge_scatter<<<cdiv((size_t)NEDGE * 32, 256), 256, 0, stream>>>(
        EX, DEN, XL, src, dst, outp, NEDGE);
    bias_act_kernel<<<cdiv((size_t)ndst * HD, 256), 256, 0, stream>>>(
        outp, bias, (size_t)ndst * HD, relu);
  };

  // Product encoder MLP.
  gemm(products, pe_W1, pe_b1, XPH, N_PRODN, PHN, DPIN, 2);   // K=769 generic
  gemm(XPH, pe_W2, pe_b2, XP, N_PRODN, HD, PHN, 0);           // K=512 vectorized

  // Layer 1 (bipartite, lazy in_channels), relu on outputs.
  gat(x_customer, DC, N_CUSTN, XP, HD, N_PRODN, ei_src, ei_dst,
      l1bp_Wl, l1bp_bl, l1bp_Wr, l1bp_br, l1bp_We, l1bp_att, l1bp_bias, HPa, 1);
  gat(XP, HD, N_PRODN, x_customer, DC, N_CUSTN, ei_dst, ei_src,
      l1pc_Wl, l1pc_bl, l1pc_Wr, l1pc_br, l1pc_We, l1pc_att, l1pc_bias, HCa, 1);

  // Layers 2 and 3 (relu after layer 2 only), ping-pong buffers.
  for (int l = 0; l < 2; ++l) {
    const float* hpin = (l == 0) ? HPa : HPb;
    const float* hcin = (l == 0) ? HCa : HCb;
    float* hpout = (l == 0) ? HPb : HPa;
    float* hcout = (l == 0) ? HCb : HCa;
    int relu = (l == 0) ? 1 : 0;
    size_t j0 = (size_t)(l * 2 + 0), j1 = (size_t)(l * 2 + 1);
    gat(hcin, HD, N_CUSTN, hpin, HD, N_PRODN, ei_src, ei_dst,
        Wl23 + j0 * HD * HD, bl23 + j0 * HD, Wr23 + j0 * HD * HD, br23 + j0 * HD,
        We23 + j0 * EDN * HD, att23 + j0 * HD, bias23 + j0 * HD, hpout, relu);
    gat(hpin, HD, N_PRODN, hcin, HD, N_CUSTN, ei_dst, ei_src,
        Wl23 + j1 * HD * HD, bl23 + j1 * HD, Wr23 + j1 * HD * HD, br23 + j1 * HD,
        We23 + j1 * EDN * HD, att23 + j1 * HD, bias23 + j1 * HD, hcout, relu);
  }

  // Fused decoder: gather-concat GEMM + relu + dot(W2) + sigmoid.
  decoder_wmma<<<cdiv(NEL, BM), dim3(128), 0, stream>>>(
      HCa, HPa, el_src, el_dst, dec_W1, dec_b1, dec_W2, dec_b2,
      (float*)d_out, NEL);
}
